// EquivariantAttention_67619965108709
// MI455X (gfx1250) — matrix operand order repaired and checked by reading, not compiled, not run on hardware
//
#include <hip/hip_runtime.h>
#include <cstdint>
#include <cstddef>

#define HEADS    8
#define DIM_HEAD 64
#define KNB      32
#define FDIM     512
#define BATCH    2
#define NPTS     2048
#define BN       (BATCH * NPTS)        // 4096 rows
#define INNER    (HEADS * DIM_HEAD)    // 512
#define NQKV     (3 * INNER)           // 1536

typedef __attribute__((ext_vector_type(16))) __bf16 v16bf;
typedef __attribute__((ext_vector_type(8)))  float  v8f;
typedef unsigned int u32x4 __attribute__((ext_vector_type(4)));
typedef int          i32x4 __attribute__((ext_vector_type(4)));
typedef int          i32x8 __attribute__((ext_vector_type(8)));

#if defined(__has_builtin)
#  if __has_builtin(__builtin_amdgcn_tensor_load_to_lds)
#    define HAVE_TDM 1
#  endif
#endif
#ifndef HAVE_TDM
#  define HAVE_TDM 0
#endif

__device__ __forceinline__ unsigned short f2bf(float f) {
  union { float f; unsigned u; } v; v.f = f;
  unsigned r = v.u + 0x7FFFu + ((v.u >> 16) & 1u);   // round-to-nearest-even
  return (unsigned short)(r >> 16);
}

// ---------------------------------------------------------------------------
// Kernel 1: fp32 -> bf16 conversions (+ transposes of B matrices so WMMA
// B-fragments / TDM panels are contiguous in K).
// ---------------------------------------------------------------------------
__global__ __launch_bounds__(256) void prep_convert(
    const float* __restrict__ feats, const float* __restrict__ w_qkv,
    const float* __restrict__ w_out,
    unsigned short* __restrict__ Abf,   // [BN, FDIM] bf16
    unsigned short* __restrict__ Bq,    // [NQKV, FDIM] bf16 (w_qkv transposed)
    unsigned short* __restrict__ Bo)    // [FDIM, INNER] bf16 (w_out transposed)
{
  int i = blockIdx.x * 256 + threadIdx.x;
  if (i < BN * FDIM) Abf[i] = f2bf(feats[i]);
  if (i < FDIM * NQKV) {
    int k = i / NQKV, n = i - k * NQKV;
    Bq[(size_t)n * FDIM + k] = f2bf(w_qkv[i]);
  }
  if (i < INNER * FDIM) {
    int k = i >> 9, n = i & 511;
    Bo[(size_t)n * INNER + k] = f2bf(w_out[i]);
  }
}

// ---------------------------------------------------------------------------
// Kernel 2: bf16 WMMA GEMM, C[m,n] = sum_k A[m,k] * Bt[n,k] (+ bias[n]).
// Workgroup tile: 256(M) x 64(N). The 64-col B panel (64 x K bf16, contiguous
// since B is pre-transposed) is staged into LDS once via the Tensor Data
// Mover (tensor_load_to_lds + s_wait_tensorcnt), then 8 waves each register-
// block 32x64: 2 A fragments (global b128) x 4 B fragments (LDS) -> 8 WMMAs
// per K-step of 32.
// ---------------------------------------------------------------------------
__global__ __launch_bounds__(256) void wmma_gemm_bf16(
    const unsigned short* __restrict__ A,   // [M,K] bf16 row-major
    const unsigned short* __restrict__ Bt,  // [N,K] bf16 row-major (B^T)
    const float* __restrict__ bias,         // [N] or nullptr
    float* __restrict__ C, int M, int N, int K)
{
  __shared__ unsigned short Bs[64 * 512];   // 64 cols x K(<=512) = 64KB

  const int wave = threadIdx.x >> 5;
  const int lane = threadIdx.x & 31;
  const int hlf  = lane >> 4;               // half-wave select
  const int idx  = lane & 15;               // A row / B column within tile
  const int m0   = blockIdx.x * 256 + wave * 32;
  const int n0   = blockIdx.y * 64;

  const unsigned short* gpanel = Bt + (size_t)n0 * K;
  const unsigned nelem = (unsigned)(64 * K);          // bf16 elems in panel

#if HAVE_TDM
  if (threadIdx.x < 32) {                   // wave 0 issues the TDM DMA
    unsigned long long ga = (unsigned long long)(uintptr_t)gpanel;
    unsigned lds_off = (unsigned)(uintptr_t)(&Bs[0]); // wg-relative LDS bytes
    u32x4 g0; i32x8 g1; i32x4 g2, g3;
    g0[0] = 1u;                                         // count=1, user D#
    g0[1] = lds_off;                                    // lds_addr
    g0[2] = (unsigned)(ga & 0xffffffffu);               // global_addr[31:0]
    g0[3] = (unsigned)((ga >> 32) & 0x01ffffffu) | (2u << 30); // [56:32]|type=2
    g1[0] = 0x00010000;                                 // data_size=1 (2B)
    g1[1] = (int)((nelem & 0xffffu) << 16);             // tensor_dim0[15:0]
    g1[2] = (int)(((nelem >> 16) & 0xffffu) | (1u << 16)); // dim0[31:16],dim1=1
    g1[3] = (int)((nelem & 0xffffu) << 16);             // tile_dim0 = nelem
    g1[4] = 1;                                          // tile_dim1 = 1
    g1[5] = (int)nelem;                                 // dim0_stride[31:0]
    g1[6] = 0; g1[7] = 0;
    g2 = (i32x4){0, 0, 0, 0};
    g3 = (i32x4){0, 0, 0, 0};
#if defined(__clang_major__) && (__clang_major__ >= 23)
    i32x8 g4 = (i32x8){0, 0, 0, 0, 0, 0, 0, 0};
    __builtin_amdgcn_tensor_load_to_lds(g0, g1, g2, g3, g4, 0);
#else
    __builtin_amdgcn_tensor_load_to_lds(g0, g1, g2, g3, 0);
#endif
    __builtin_amdgcn_s_wait_tensorcnt(0);
  }
#else
  {
    const uint4* src = (const uint4*)gpanel;
    uint4* dst = (uint4*)Bs;
    const int total = (64 * K) / 8;                     // uint4 = 8 bf16
    for (int t = threadIdx.x; t < total; t += 256) dst[t] = src[t];
  }
#endif
  __syncthreads();

  const unsigned short* ap0 = A + (size_t)(m0 + idx) * K + hlf * 8;
  const unsigned short* ap1 = ap0 + (size_t)16 * K;
  const unsigned short* bsp = Bs + (size_t)idx * K + hlf * 8;

  v8f acc[2][4] = {};
  for (int kb = 0; kb < K; kb += 32) {
    v16bf a0, a1;
    // A: elems 0..7 -> K = kb + hlf*8 ; elems 8..15 -> K = kb + 16 + hlf*8
    ((uint4*)&a0)[0] = *(const uint4*)(ap0 + kb);
    ((uint4*)&a0)[1] = *(const uint4*)(ap0 + kb + 16);
    ((uint4*)&a1)[0] = *(const uint4*)(ap1 + kb);
    ((uint4*)&a1)[1] = *(const uint4*)(ap1 + kb + 16);
#pragma unroll
    for (int t = 0; t < 4; ++t) {
      // B: elems 0..7 -> K = kb + hlf*8 ; elems 8..15 -> K = kb + 16 + hlf*8
      v16bf bfrag;
      ((uint4*)&bfrag)[0] = *(const uint4*)(bsp + (size_t)t * 16 * K + kb);
      ((uint4*)&bfrag)[1] = *(const uint4*)(bsp + (size_t)t * 16 * K + kb + 16);
      acc[0][t] = __builtin_amdgcn_wmma_f32_16x16x32_bf16(
          false, a0, false, bfrag, (short)0, acc[0][t], false, false);
      acc[1][t] = __builtin_amdgcn_wmma_f32_16x16x32_bf16(
          false, a1, false, bfrag, (short)0, acc[1][t], false, false);
    }
  }

#pragma unroll
  for (int r = 0; r < 2; ++r) {
#pragma unroll
    for (int t = 0; t < 4; ++t) {
      const int col = n0 + t * 16 + idx;
      const float bv = bias ? bias[col] : 0.0f;
      float* crow = C + (size_t)(m0 + r * 16 + hlf * 8) * N + col;
#pragma unroll
      for (int v = 0; v < 8; ++v) crow[(size_t)v * N] = acc[r][t][v] + bv;
    }
  }
}

// ---------------------------------------------------------------------------
// Kernel 3: per-query 32-nearest-neighbor selection (iterative argmin in LDS).
// ---------------------------------------------------------------------------
__global__ __launch_bounds__(128) void topk_kernel(
    const float* __restrict__ coors, int* __restrict__ nbhd,
    float* __restrict__ ndist, float* __restrict__ ncoor)
{
  const int bi = blockIdx.x;           // [0, BN)
  const int bb = bi / NPTS;
  const int tid = threadIdx.x;
  __shared__ float d2[NPTS];
  __shared__ float rbest[128];
  __shared__ int   rbj[128];

  const float qx = coors[(size_t)bi * 3 + 0];
  const float qy = coors[(size_t)bi * 3 + 1];
  const float qz = coors[(size_t)bi * 3 + 2];
  const float* cb = coors + (size_t)bb * NPTS * 3;

  for (int j = tid; j < NPTS; j += 128) {
    float dx = qx - cb[j * 3 + 0];
    float dy = qy - cb[j * 3 + 1];
    float dz = qz - cb[j * 3 + 2];
    d2[j] = dx * dx + dy * dy + dz * dz;
  }
  __syncthreads();

  for (int s = 0; s < KNB; ++s) {
    float best = 3.4e38f; int bj = NPTS;
    for (int j = tid; j < NPTS; j += 128) {
      float v = d2[j];
      if (v < best) { best = v; bj = j; }
    }
    rbest[tid] = best; rbj[tid] = bj;
    __syncthreads();
    for (int off = 64; off > 0; off >>= 1) {
      if (tid < off) {
        float o = rbest[tid + off];
        if (o < rbest[tid] || (o == rbest[tid] && rbj[tid + off] < rbj[tid])) {
          rbest[tid] = o; rbj[tid] = rbj[tid + off];
        }
      }
      __syncthreads();
    }
    if (tid == 0) {
      int j = rbj[0];
      nbhd[(size_t)bi * KNB + s]  = j;
      ndist[(size_t)bi * KNB + s] = sqrtf(rbest[0]);
      ncoor[((size_t)bi * KNB + s) * 3 + 0] = qx - cb[j * 3 + 0];
      ncoor[((size_t)bi * KNB + s) * 3 + 1] = qy - cb[j * 3 + 1];
      ncoor[((size_t)bi * KNB + s) * 3 + 2] = qz - cb[j * 3 + 2];
      d2[j] = 3.4e38f;
    }
    __syncthreads();
  }
}

// ---------------------------------------------------------------------------
// Kernel 4: per-query attention core (rotary, logits, tiny MLP, softmaxes,
// coord output, attn*v_rot staged as bf16 for the final WMMA GEMM).
// ---------------------------------------------------------------------------
__global__ __launch_bounds__(256) void attn_kernel(
    const float* __restrict__ QKV, const int* __restrict__ nbhd,
    const float* __restrict__ ndist, const float* __restrict__ ncoor,
    const float* __restrict__ wc1, const float* __restrict__ bc1,
    const float* __restrict__ wc2, const float* __restrict__ bc2,
    const float* __restrict__ wg,  const float* __restrict__ bg,
    const float* __restrict__ coors_scale, const float* __restrict__ coors_combine,
    unsigned short* __restrict__ OutA, float* __restrict__ coors_out)
{
  const int bi = blockIdx.x;           // [0, BN)
  const int bb = bi / NPTS;
  const int tid = threadIdx.x;

  __shared__ float q_s[INNER];
  __shared__ float cos_s[KNB * 32];
  __shared__ float sin_s[KNB * 32];
  __shared__ float qk_s[HEADS * KNB];
  __shared__ float cw_s[HEADS * KNB];
  __shared__ float gate_s[HEADS * KNB];
  __shared__ float catt_s[HEADS * KNB];
  __shared__ float att_s[HEADS * KNB];
  __shared__ float dist_s[KNB];
  __shared__ float rc_s[KNB * 3];
  __shared__ float rnv_s[KNB];
  __shared__ int   nb_s[KNB];

  const float* myrow = QKV + (size_t)bi * NQKV;
  for (int t = tid; t < INNER; t += 256) q_s[t] = myrow[t];  // q (rotary id.)
  if (tid < KNB) {
    nb_s[tid]   = nbhd[(size_t)bi * KNB + tid];
    dist_s[tid] = ndist[(size_t)bi * KNB + tid];
  }
  if (tid < KNB * 3) rc_s[tid] = ncoor[(size_t)bi * KNB * 3 + tid];
  __syncthreads();

  // rotary tables: f[j][fi] = min(100*d, 5000) / 10000^(2fi/64)
  for (int t = tid; t < KNB * 32; t += 256) {
    int j = t >> 5, fi = t & 31;
    float td   = fminf(dist_s[j] * 100.0f, 5000.0f);
    float invf = exp2f(-13.287712379549449f * (float)(2 * fi) * (1.0f / 64.0f));
    float f = td * invf, sv, cv;
    sincosf(f, &sv, &cv);
    cos_s[t] = cv; sin_s[t] = sv;
  }
  if (tid < KNB) {
    float rx = rc_s[tid * 3], ry = rc_s[tid * 3 + 1], rz = rc_s[tid * 3 + 2];
    float nrm = sqrtf(rx * rx + ry * ry + rz * rz);
    rnv_s[tid] = coors_scale[0] / fmaxf(nrm, 1e-8f);
  }
  __syncthreads();

  // qk logits: one thread per (j, h)
  {
    const int j = tid >> 3, h = tid & 7;
    const float* kr = QKV + (size_t)(bb * NPTS + nb_s[j]) * NQKV + INNER + h * DIM_HEAD;
    float kl[DIM_HEAD];
#pragma unroll
    for (int d = 0; d < DIM_HEAD; ++d) kl[d] = kr[d];
    const float* qh = q_s + h * DIM_HEAD;
    const float* cj = cos_s + j * 32;
    const float* sj = sin_s + j * 32;
    float acc = 0.0f;
#pragma unroll
    for (int p = 0; p < 32; ++p)             // rot[p] = k[p]c - k[2p+1]s
      acc += qh[p] * (kl[p] * cj[p >> 1] - kl[2 * p + 1] * sj[p >> 1]);
#pragma unroll
    for (int p = 32; p < 64; ++p)            // rot[p] = k[p]c + k[2p-64]s
      acc += qh[p] * (kl[p] * cj[p >> 1] + kl[2 * p - 64] * sj[p >> 1]);
    qk_s[h * KNB + j] = acc * 0.125f;        // SCALE = 64^-0.5
  }
  __syncthreads();

  // tiny MLP (8->16 gelu ->8) and tanh gate, one thread per neighbor j
  if (tid < KNB) {
    const int j = tid;
    float hid[16];
#pragma unroll
    for (int m = 0; m < 16; ++m) {
      float x = bc1[m];
#pragma unroll
      for (int h = 0; h < HEADS; ++h) x += qk_s[h * KNB + j] * wc1[h * 16 + m];
      hid[m] = 0.5f * x * (1.0f + tanhf(0.7978845608028654f * (x + 0.044715f * x * x * x)));
    }
#pragma unroll
    for (int h = 0; h < HEADS; ++h) {
      float cw = bc2[h];
#pragma unroll
      for (int m = 0; m < 16; ++m) cw += hid[m] * wc2[m * HEADS + h];
      cw_s[h * KNB + j] = cw;
      float g = bg[h];
#pragma unroll
      for (int hp = 0; hp < HEADS; ++hp) g += qk_s[hp * KNB + j] * wg[hp * HEADS + h];
      gate_s[h * KNB + j] = tanhf(g);
    }
  }
  __syncthreads();

  // softmaxes over neighbors, one thread per head
  if (tid < HEADS) {
    const int h = tid;
    float mx = -3.4e38f, mq = -3.4e38f;
    for (int j = 0; j < KNB; ++j) {
      mx = fmaxf(mx, cw_s[h * KNB + j]);
      mq = fmaxf(mq, qk_s[h * KNB + j]);
    }
    float se = 0.0f, sq = 0.0f;
    for (int j = 0; j < KNB; ++j) {
      float e  = expf(cw_s[h * KNB + j] - mx); catt_s[h * KNB + j] = e;  se += e;
      float e2 = expf(qk_s[h * KNB + j] - mq); att_s[h * KNB + j]  = e2; sq += e2;
    }
    float ise = 1.0f / se, isq = 1.0f / sq;
    for (int j = 0; j < KNB; ++j) { catt_s[h * KNB + j] *= ise; att_s[h * KNB + j] *= isq; }
  }
  __syncthreads();

  // coordinate output
  if (tid < 3) {
    const int c = tid;
    float acc = 0.0f;
    for (int h = 0; h < HEADS; ++h) {
      float rel = 0.0f;
      for (int j = 0; j < KNB; ++j)
        rel += catt_s[h * KNB + j] * rc_s[j * 3 + c] * rnv_s[j] * gate_s[h * KNB + j];
      acc += rel * coors_combine[h];
    }
    coors_out[(size_t)bi * 3 + c] = acc;
  }

  // feature output: out[h,d] = sum_j attn[h,j] * rot(v)[j,h,d]; 2 (h,d)/thread
#pragma unroll
  for (int rep = 0; rep < 2; ++rep) {
    const int p = tid + rep * 256;
    const int h = p >> 6, d = p & 63;
    float o = 0.0f;
    for (int j = 0; j < KNB; ++j) {
      const float* vr = QKV + (size_t)(bb * NPTS + nb_s[j]) * NQKV + 2 * INNER + h * DIM_HEAD;
      float c  = cos_s[j * 32 + (d >> 1)];
      float s  = sin_s[j * 32 + (d >> 1)];
      float vv = vr[d];
      float pr = (d < 32) ? -vr[2 * d + 1] : vr[2 * d - 64];
      o += att_s[h * KNB + j] * (vv * c + pr * s);
    }
    OutA[(size_t)bi * INNER + p] = f2bf(o);
  }
}

// ---------------------------------------------------------------------------
extern "C" void kernel_launch(void* const* d_in, const int* in_sizes, int n_in,
                              void* d_out, int out_size, void* d_ws, size_t ws_size,
                              hipStream_t stream)
{
  const float* feats  = (const float*)d_in[0];
  const float* coors  = (const float*)d_in[1];
  const float* w_qkv  = (const float*)d_in[2];
  const float* w_out  = (const float*)d_in[3];
  const float* b_out  = (const float*)d_in[4];
  const float* wc1    = (const float*)d_in[5];
  const float* bc1    = (const float*)d_in[6];
  const float* wc2    = (const float*)d_in[7];
  const float* bc2    = (const float*)d_in[8];
  const float* wg     = (const float*)d_in[9];
  const float* bg     = (const float*)d_in[10];
  const float* cscale = (const float*)d_in[11];
  const float* ccomb  = (const float*)d_in[12];
  (void)in_sizes; (void)n_in; (void)out_size; (void)ws_size;

  char* ws = (char*)d_ws;
  size_t off = 0;
  auto take = [&](size_t bytes) -> char* {
    char* p = ws + off;
    off += (bytes + 255) & ~(size_t)255;
    return p;
  };
  unsigned short* Abf  = (unsigned short*)take((size_t)BN * FDIM * 2);
  unsigned short* Bq   = (unsigned short*)take((size_t)NQKV * FDIM * 2);
  unsigned short* Bo   = (unsigned short*)take((size_t)FDIM * INNER * 2);
  float*          QKV  = (float*)take((size_t)BN * NQKV * 4);
  int*            NB   = (int*)take((size_t)BN * KNB * 4);
  float*          ND   = (float*)take((size_t)BN * KNB * 4);
  float*          NC   = (float*)take((size_t)BN * KNB * 3 * 4);
  unsigned short* OutA = (unsigned short*)take((size_t)BN * INNER * 2);

  float* out_feats = (float*)d_out;                 // [BN, 512]
  float* out_coors = out_feats + (size_t)BN * FDIM; // [BN, 3]

  prep_convert<<<(BN * FDIM + 255) / 256, 256, 0, stream>>>(
      feats, w_qkv, w_out, Abf, Bq, Bo);

  // QKV GEMM: [4096,512] x [512,1536], 256x64 workgroup tiles
  wmma_gemm_bf16<<<dim3(BN / 256, NQKV / 64), 256, 0, stream>>>(
      Abf, Bq, nullptr, QKV, BN, NQKV, FDIM);

  topk_kernel<<<BN, 128, 0, stream>>>(coors, NB, ND, NC);

  attn_kernel<<<BN, 256, 0, stream>>>(
      QKV, NB, ND, NC, wc1, bc1, wc2, bc2, wg, bg, cscale, ccomb, OutA, out_coors);

  // Output GEMM: [4096,512] x [512,512] + b_out
  wmma_gemm_bf16<<<dim3(BN / 256, FDIM / 64), 256, 0, stream>>>(
      OutA, Bo, b_out, out_feats, BN, FDIM, INNER);
}
